// KG_GAT_82240033784432
// MI455X (gfx1250) — compile-verified, matrix operand order checked by reading, not run on hardware
//
#include <hip/hip_runtime.h>
#include <stdint.h>

// Problem constants (shapes fixed by the reference).
#define IN_F   768
#define HID_F  256
#define OUT_F  128
#define HEADS  8
#define NREL   6
#define RDIM   64
#define AST    40   // padded LDS row stride (elements) for 32-elem rows: 80B, 16B-aligned

typedef __attribute__((ext_vector_type(16))) __bf16 v16bf;
typedef __attribute__((ext_vector_type(8)))  float  v8f;

// ---------- helpers ----------
__device__ __forceinline__ unsigned short f2bf(float f) {
    unsigned u = __float_as_uint(f);
    u += 0x7fffu + ((u >> 16) & 1u);           // round-to-nearest-even
    return (unsigned short)(u >> 16);
}
__device__ __forceinline__ unsigned pack2(float a, float b) {
    return (unsigned)f2bf(a) | ((unsigned)f2bf(b) << 16);
}
// monotone float->uint encoding for atomicMax on floats
__device__ __forceinline__ unsigned fkey(float x) {
    unsigned u = __float_as_uint(x);
    return (u & 0x80000000u) ? ~u : (u | 0x80000000u);
}
__device__ __forceinline__ float fdec(unsigned k) {
    unsigned u = (k & 0x80000000u) ? (k & 0x7fffffffu) : ~k;
    return __uint_as_float(u);
}

// async 16B copy global -> LDS (gfx1250: GLOBAL_LOAD_ASYNC_TO_LDS_B128, ASYNCcnt)
__device__ __forceinline__ void cp_async16(const unsigned short* g, const unsigned short* l) {
    // generic shared pointers carry the LDS offset in bits [31:0]
    unsigned loff = (unsigned)(size_t)l;
    asm volatile("global_load_async_to_lds_b128 %0, %1, off"
                 :: "v"(loff), "v"(g) : "memory");
}
__device__ __forceinline__ void wait_async_le3() {
    asm volatile("s_wait_asynccnt 0x3" ::: "memory");
}

// ---------- utility kernels ----------
__global__ void k_zero(unsigned* __restrict__ p, long long nwords) {
    long long i = (long long)blockIdx.x * 256 + threadIdx.x;
    if (i < nwords) p[i] = 0u;
}

// fp32 -> bf16, 8 elements/thread
__global__ void k_cast8(const float* __restrict__ src, unsigned short* __restrict__ dst,
                        long long n8) {
    long long i = (long long)blockIdx.x * 256 + threadIdx.x;
    if (i >= n8) return;
    const float* s = src + i * 8;
    uint4 o;
    o.x = pack2(s[0], s[1]); o.y = pack2(s[2], s[3]);
    o.z = pack2(s[4], s[5]); o.w = pack2(s[6], s[7]);
    *(uint4*)(dst + i * 8) = o;
}

// transpose-cast: src [KO, NO] fp32 row-major -> dst [NO, KO] bf16 (i.e. B^T)
__global__ void k_castT(const float* __restrict__ src, unsigned short* __restrict__ dst,
                        int NO, int KO) {
    int t = blockIdx.x * 256 + threadIdx.x;
    if (t >= NO * KO) return;
    int nn = t / KO, kk = t % KO;
    dst[t] = f2bf(src[(long long)kk * NO + nn]);
}

// ---------- bf16 WMMA GEMM, async-LDS double-buffered ----------
// C[M,N] = A[M,K] * B[K,N] with Bt = B^T row-major (bf16).
// block = 256 threads (8 waves), block tile 64(M) x 128(N), wave tile 32x32.
// Per k-stage (K chunk of 32): A tile 64x32, B tile 128x32 staged via
// global_load_async_to_lds_b128; two stages in flight (ASYNCcnt pipelining).
__global__ __launch_bounds__(256) void k_gemm_bf16(
    const unsigned short* __restrict__ A, const unsigned short* __restrict__ Bt,
    float* __restrict__ C, int M, int Ncols, int K) {
    __shared__ unsigned short sA[2][64 * AST];
    __shared__ unsigned short sB[2][128 * AST];

    int t = threadIdx.x;
    int lane = t & 31, wave = t >> 5;
    int wm = wave >> 2, wn = wave & 3;
    int rowBlk = blockIdx.x * 64, colBlk = blockIdx.y * 128;
    int m16 = lane & 15, half = lane >> 4;

    // per-thread copy assignments:
    // A: 64 rows x 4 segs of 8 elems  -> 256 copies, 1/thread
    // B: 128 rows x 4 segs            -> 512 copies, 2/thread
    int arow = t >> 2, aseg = t & 3;
    int ar = rowBlk + arow; if (ar >= M) ar = M - 1;     // clamp; stores guarded later
    const long long aOff  = (long long)ar * K + aseg * 8;
    const long long bOff0 = (long long)(colBlk + (t >> 2)) * K + (t & 3) * 8;
    const long long bOff1 = (long long)(colBlk + 64 + (t >> 2)) * K + (t & 3) * 8;
    const int aDst  = arow * AST + aseg * 8;
    const int bDst0 = (t >> 2) * AST + (t & 3) * 8;
    const int bDst1 = (64 + (t >> 2)) * AST + (t & 3) * 8;

    auto issue = [&](int buf, int k0) {
        cp_async16(A + aOff + k0, &sA[buf][aDst]);
        cp_async16(Bt + bOff0 + k0, &sB[buf][bDst0]);
        cp_async16(Bt + bOff1 + k0, &sB[buf][bDst1]);
    };

    v8f acc[2][2];
#pragma unroll
    for (int i = 0; i < 2; i++)
#pragma unroll
        for (int j = 0; j < 2; j++) acc[i][j] = (v8f){0, 0, 0, 0, 0, 0, 0, 0};

    const int nk = K >> 5;
    issue(0, 0);
    issue(1, (nk > 1 ? 32 : 0));

    for (int ks = 0; ks < nk; ks++) {
        int buf = ks & 1;
        wait_async_le3();        // oldest in-flight stage (== stage ks) retired
        __syncthreads();         // block-wide visibility of LDS tiles

        union { v16bf v; uint4 q[2]; } af[2], bf[2];
#pragma unroll
        for (int i = 0; i < 2; i++) {
            const unsigned short* p = &sA[buf][(wm * 32 + i * 16 + m16) * AST + half * 8];
            af[i].q[0] = *(const uint4*)p;          // K in [half*8, half*8+8)
            af[i].q[1] = *(const uint4*)(p + 16);   // K in [16+half*8, ...)
        }
#pragma unroll
        for (int j = 0; j < 2; j++) {
            const unsigned short* p = &sB[buf][(wn * 32 + j * 16 + m16) * AST + half * 8];
            bf[j].q[0] = *(const uint4*)p;
            bf[j].q[1] = *(const uint4*)(p + 16);
        }
#pragma unroll
        for (int i = 0; i < 2; i++)
#pragma unroll
            for (int j = 0; j < 2; j++)
                acc[i][j] = __builtin_amdgcn_wmma_f32_16x16x32_bf16(
                    false, af[i].v, false, bf[j].v, (short)0, acc[i][j], false, false);

        __syncthreads();         // everyone done reading buf before refill
        int knext = ks + 2;
        int k0 = (knext < nk ? knext : nk - 1) * 32;   // tail: dummy refill, never read
        issue(buf, k0);
    }

#pragma unroll
    for (int i = 0; i < 2; i++)
#pragma unroll
        for (int j = 0; j < 2; j++) {
            int col = colBlk + wn * 32 + j * 16 + m16;
#pragma unroll
            for (int v = 0; v < 8; v++) {
                int row = rowBlk + wm * 32 + i * 16 + half * 8 + v;
                if (row < M) C[(long long)row * Ncols + col] = acc[i][j][v];
            }
        }
}

// ---------- rel_alpha precompute: rel_alpha[r][h] = rel_emb[r,:] . V[:,h] ----------
__global__ void k_prep(const float* __restrict__ We1, const float* __restrict__ att_e1,
                       const float* __restrict__ rel_emb, float* __restrict__ rel_alpha) {
    __shared__ float Vs[RDIM * HEADS];
    int t = threadIdx.x;                      // 512 = RDIM*HEADS
    int d = t >> 3, h = t & 7;
    float s = 0.f;
    for (int c = 0; c < 32; c++) s += We1[d * HID_F + h * 32 + c] * att_e1[h * 32 + c];
    Vs[d * HEADS + h] = s;
    __syncthreads();
    if (t < NREL * HEADS) {
        int r = t >> 3, hh = t & 7;
        float a = 0.f;
        for (int dd = 0; dd < RDIM; dd++) a += rel_emb[r * RDIM + dd] * Vs[dd * HEADS + hh];
        rel_alpha[t] = a;
    }
}

// per-(node,relation) incoming edge counts
__global__ void k_cnt(const int* __restrict__ dst0, const int* __restrict__ et,
                      int* __restrict__ cnt, int E) {
    int e = blockIdx.x * 256 + threadIdx.x;
    if (e >= E) return;
    atomicAdd(&cnt[(long long)dst0[e] * NREL + et[e]], 1);
}

// layer-1 per-node logits + self-loop edge-attr logit
__global__ void k_nodeatt1(const float* __restrict__ h1, const float* __restrict__ as,
                           const float* __restrict__ ad, const int* __restrict__ cnt,
                           const float* __restrict__ rel_alpha, float* __restrict__ al_s,
                           float* __restrict__ al_d, float* __restrict__ sa, int n) {
    int t = blockIdx.x * 256 + threadIdx.x;
    if (t >= n * HEADS) return;
    int i = t >> 3, h = t & 7;
    const float* row = h1 + (long long)i * HID_F + h * 32;
    float s1 = 0.f, s2 = 0.f;
    for (int c = 0; c < 32; c++) {
        float v = row[c];
        s1 += v * as[h * 32 + c];
        s2 += v * ad[h * 32 + c];
    }
    al_s[t] = s1; al_d[t] = s2;
    float tot = 0.f, acc = 0.f;
    for (int r = 0; r < NREL; r++) {
        float cr = (float)cnt[(long long)i * NREL + r];
        tot += cr;
        acc += cr * rel_alpha[r * HEADS + h];
    }
    sa[t] = acc / fmaxf(tot, 1.f);
}

// ---------- layer-1 segment softmax passes ----------
__global__ void k_amax1(const int* __restrict__ srcE, const int* __restrict__ dstE,
                        const int* __restrict__ et, const float* __restrict__ al_s,
                        const float* __restrict__ al_d, const float* __restrict__ rel_alpha,
                        const float* __restrict__ sa, float* __restrict__ exbuf,
                        unsigned* __restrict__ mx, int E, int Etot) {
    int t = blockIdx.x * 256 + threadIdx.x;
    if (t >= Etot * HEADS) return;
    int e = t >> 3, h = t & 7;
    int s, d; float extra;
    if (e < E) { s = srcE[e]; d = dstE[e]; extra = rel_alpha[et[e] * HEADS + h]; }
    else       { s = d = e - E;            extra = sa[(long long)s * HEADS + h]; }
    float a = al_s[(long long)s * HEADS + h] + al_d[(long long)d * HEADS + h] + extra;
    a = a > 0.f ? a : 0.2f * a;            // leaky_relu(0.2)
    exbuf[t] = a;
    atomicMax(&mx[(long long)d * HEADS + h], fkey(a));
}

__global__ void k_aexp1(const int* __restrict__ dstE, float* __restrict__ exbuf,
                        const unsigned* __restrict__ mx, float* __restrict__ den,
                        int E, int Etot) {
    int t = blockIdx.x * 256 + threadIdx.x;
    if (t >= Etot * HEADS) return;
    int e = t >> 3, h = t & 7;
    int d = (e < E) ? dstE[e] : (e - E);
    float ex = __expf(exbuf[t] - fdec(mx[(long long)d * HEADS + h]));
    exbuf[t] = ex;
    atomicAdd(&den[(long long)d * HEADS + h], ex);
}

// weighted scatter-add, one wave per edge (256 channels, 8 per lane)
__global__ __launch_bounds__(256) void k_agg1(const int* __restrict__ srcE,
                                              const int* __restrict__ dstE,
                                              const float* __restrict__ h1,
                                              const float* __restrict__ exbuf,
                                              const float* __restrict__ den,
                                              float* __restrict__ out1, int E, int Etot) {
    int e = blockIdx.x * 8 + (threadIdx.x >> 5);
    if (e >= Etot) return;
    int lane = threadIdx.x & 31;
    int s, d;
    if (e < E) { s = srcE[e]; d = dstE[e]; } else { s = d = e - E; }
    int h = lane >> 2;
    float w = exbuf[(long long)e * HEADS + h] / (den[(long long)d * HEADS + h] + 1e-16f);
    const float4* sp = (const float4*)(h1 + (long long)s * HID_F) + lane * 2;
    float4 a = sp[0], b = sp[1];
    float* o = out1 + (long long)d * HID_F + lane * 8;
    atomicAdd(o + 0, a.x * w); atomicAdd(o + 1, a.y * w);
    atomicAdd(o + 2, a.z * w); atomicAdd(o + 3, a.w * w);
    atomicAdd(o + 4, b.x * w); atomicAdd(o + 5, b.y * w);
    atomicAdd(o + 6, b.z * w); atomicAdd(o + 7, b.w * w);
}

// +bias, LayerNorm, ELU -> bf16; one wave per node (256 ch)
__global__ __launch_bounds__(256) void k_lnelu1(const float* __restrict__ out1,
                                                const float* __restrict__ b1,
                                                const float* __restrict__ g1,
                                                const float* __restrict__ be1,
                                                unsigned short* __restrict__ hact, int n) {
    int i = blockIdx.x * 8 + (threadIdx.x >> 5);
    if (i >= n) return;
    int lane = threadIdx.x & 31;
    const float4* p = (const float4*)(out1 + (long long)i * HID_F) + lane * 2;
    float4 a = p[0], b = p[1];
    float x[8] = {a.x, a.y, a.z, a.w, b.x, b.y, b.z, b.w};
    float s = 0.f, q = 0.f;
#pragma unroll
    for (int j = 0; j < 8; j++) {
        x[j] += b1[lane * 8 + j];
        s += x[j]; q += x[j] * x[j];
    }
    for (int off = 16; off; off >>= 1) { s += __shfl_xor(s, off); q += __shfl_xor(q, off); }
    float m = s * (1.f / HID_F);
    float inv = rsqrtf(q * (1.f / HID_F) - m * m + 1e-5f);
    float y[8];
#pragma unroll
    for (int j = 0; j < 8; j++) {
        int c = lane * 8 + j;
        float v = (x[j] - m) * inv * g1[c] + be1[c];
        y[j] = v > 0.f ? v : (__expf(v) - 1.f);    // ELU(alpha=1)
    }
    uint4 o;
    o.x = pack2(y[0], y[1]); o.y = pack2(y[2], y[3]);
    o.z = pack2(y[4], y[5]); o.w = pack2(y[6], y[7]);
    *(uint4*)(hact + (long long)i * HID_F + lane * 8) = o;
}

// layer-2 per-node logits (1 head, 128 ch); one wave per node
__global__ __launch_bounds__(256) void k_nodeatt2(const float* __restrict__ h2,
                                                  const float* __restrict__ as2,
                                                  const float* __restrict__ ad2,
                                                  float* __restrict__ al_s,
                                                  float* __restrict__ al_d, int n) {
    int i = blockIdx.x * 8 + (threadIdx.x >> 5);
    if (i >= n) return;
    int lane = threadIdx.x & 31;
    float4 v = *((const float4*)(h2 + (long long)i * OUT_F) + lane);
    float4 a = *((const float4*)as2 + lane);
    float4 b = *((const float4*)ad2 + lane);
    float s1 = v.x * a.x + v.y * a.y + v.z * a.z + v.w * a.w;
    float s2 = v.x * b.x + v.y * b.y + v.z * b.z + v.w * b.w;
    for (int off = 16; off; off >>= 1) { s1 += __shfl_xor(s1, off); s2 += __shfl_xor(s2, off); }
    if (lane == 0) { al_s[i] = s1; al_d[i] = s2; }
}

__global__ void k_amax2(const int* __restrict__ srcE, const int* __restrict__ dstE,
                        const float* __restrict__ al_s, const float* __restrict__ al_d,
                        float* __restrict__ exbuf, unsigned* __restrict__ mx, int E, int Etot) {
    int e = blockIdx.x * 256 + threadIdx.x;
    if (e >= Etot) return;
    int s, d;
    if (e < E) { s = srcE[e]; d = dstE[e]; } else { s = d = e - E; }
    float a = al_s[s] + al_d[d];
    a = a > 0.f ? a : 0.2f * a;
    exbuf[e] = a;
    atomicMax(&mx[d], fkey(a));
}

__global__ void k_aexp2(const int* __restrict__ dstE, float* __restrict__ exbuf,
                        const unsigned* __restrict__ mx, float* __restrict__ den,
                        int E, int Etot) {
    int e = blockIdx.x * 256 + threadIdx.x;
    if (e >= Etot) return;
    int d = (e < E) ? dstE[e] : (e - E);
    float ex = __expf(exbuf[e] - fdec(mx[d]));
    exbuf[e] = ex;
    atomicAdd(&den[d], ex);
}

__global__ __launch_bounds__(256) void k_agg2(const int* __restrict__ srcE,
                                              const int* __restrict__ dstE,
                                              const float* __restrict__ h2,
                                              const float* __restrict__ exbuf,
                                              const float* __restrict__ den,
                                              float* __restrict__ out2, int E, int Etot) {
    int e = blockIdx.x * 8 + (threadIdx.x >> 5);
    if (e >= Etot) return;
    int lane = threadIdx.x & 31;
    int s, d;
    if (e < E) { s = srcE[e]; d = dstE[e]; } else { s = d = e - E; }
    float w = exbuf[e] / (den[d] + 1e-16f);
    float4 v = *((const float4*)(h2 + (long long)s * OUT_F) + lane);
    float* o = out2 + (long long)d * OUT_F + lane * 4;
    atomicAdd(o + 0, v.x * w); atomicAdd(o + 1, v.y * w);
    atomicAdd(o + 2, v.z * w); atomicAdd(o + 3, v.w * w);
}

// final +bias, LayerNorm -> d_out; one wave per node (128 ch)
__global__ __launch_bounds__(256) void k_ln2(const float* __restrict__ out2,
                                             const float* __restrict__ b2,
                                             const float* __restrict__ g2,
                                             const float* __restrict__ be2,
                                             float* __restrict__ outp, int n) {
    int i = blockIdx.x * 8 + (threadIdx.x >> 5);
    if (i >= n) return;
    int lane = threadIdx.x & 31;
    float4 v = *((const float4*)(out2 + (long long)i * OUT_F) + lane);
    float x[4] = {v.x, v.y, v.z, v.w};
    float s = 0.f, q = 0.f;
#pragma unroll
    for (int j = 0; j < 4; j++) {
        x[j] += b2[lane * 4 + j];
        s += x[j]; q += x[j] * x[j];
    }
    for (int off = 16; off; off >>= 1) { s += __shfl_xor(s, off); q += __shfl_xor(q, off); }
    float m = s * (1.f / OUT_F);
    float inv = rsqrtf(q * (1.f / OUT_F) - m * m + 1e-5f);
    float4 o;
    o.x = (x[0] - m) * inv * g2[lane * 4 + 0] + be2[lane * 4 + 0];
    o.y = (x[1] - m) * inv * g2[lane * 4 + 1] + be2[lane * 4 + 1];
    o.z = (x[2] - m) * inv * g2[lane * 4 + 2] + be2[lane * 4 + 2];
    o.w = (x[3] - m) * inv * g2[lane * 4 + 3] + be2[lane * 4 + 3];
    *((float4*)(outp + (long long)i * OUT_F) + lane) = o;
}

// ---------- host launcher ----------
static inline int cdiv(long long a, long long b) { return (int)((a + b - 1) / b); }

extern "C" void kernel_launch(void* const* d_in, const int* in_sizes, int n_in,
                              void* d_out, int out_size, void* d_ws, size_t ws_size,
                              hipStream_t stream) {
    const float* x        = (const float*)d_in[0];
    const int*   eidx     = (const int*)d_in[1];
    const int*   etype    = (const int*)d_in[2];
    const float* rel_emb  = (const float*)d_in[3];
    const float* W1       = (const float*)d_in[4];
    const float* att_src1 = (const float*)d_in[5];
    const float* att_dst1 = (const float*)d_in[6];
    const float* We1      = (const float*)d_in[7];
    const float* att_e1   = (const float*)d_in[8];
    const float* b1       = (const float*)d_in[9];
    const float* g1       = (const float*)d_in[10];
    const float* beta1    = (const float*)d_in[11];
    const float* W2       = (const float*)d_in[12];
    const float* att_src2 = (const float*)d_in[13];
    const float* att_dst2 = (const float*)d_in[14];
    const float* b2       = (const float*)d_in[15];
    const float* g2       = (const float*)d_in[16];
    const float* beta2    = (const float*)d_in[17];

    const int n    = in_sizes[0] / IN_F;   // 50000
    const int E    = in_sizes[2];          // 800000
    const int Etot = E + n;
    const int* srcE = eidx;
    const int* dstE = eidx + E;

    // workspace layout (256B aligned slabs)
    char* base = (char*)d_ws;
    size_t off = 0;
    auto alloc = [&](size_t bytes) -> char* {
        char* p = base + off;
        off = (off + bytes + 255) & ~(size_t)255;
        return p;
    };
    unsigned short* xb    = (unsigned short*)alloc((size_t)n * IN_F * 2);
    unsigned short* W1t   = (unsigned short*)alloc((size_t)IN_F * HID_F * 2);
    unsigned short* W2t   = (unsigned short*)alloc((size_t)HID_F * OUT_F * 2);
    unsigned short* hact  = (unsigned short*)alloc((size_t)n * HID_F * 2);
    float* h1       = (float*)alloc((size_t)n * HID_F * 4);
    float* h2       = (float*)alloc((size_t)n * OUT_F * 4);
    float* al_s1    = (float*)alloc((size_t)n * HEADS * 4);
    float* al_d1    = (float*)alloc((size_t)n * HEADS * 4);
    float* sa1      = (float*)alloc((size_t)n * HEADS * 4);
    float* rel_alpha= (float*)alloc((size_t)NREL * HEADS * 4);
    float* al_s2    = (float*)alloc((size_t)n * 4);
    float* al_d2    = (float*)alloc((size_t)n * 4);
    float* ex1      = (float*)alloc((size_t)Etot * HEADS * 4);
    float* ex2      = (float*)alloc((size_t)Etot * 4);
    // --- contiguous zero-initialized span ---
    char* zbeg = base + off;
    float*    out1 = (float*)alloc((size_t)n * HID_F * 4);
    float*    out2 = (float*)alloc((size_t)n * OUT_F * 4);
    float*    den1 = (float*)alloc((size_t)n * HEADS * 4);
    float*    den2 = (float*)alloc((size_t)n * 4);
    unsigned* mx1  = (unsigned*)alloc((size_t)n * HEADS * 4);
    unsigned* mx2  = (unsigned*)alloc((size_t)n * 4);
    int*      cnt  = (int*)alloc((size_t)n * NREL * 4);
    long long zwords = (long long)((base + off) - zbeg) / 4;

    // 0) zero the accumulator span (mx key 0 encodes <= -inf, valid for max)
    k_zero<<<cdiv(zwords, 256), 256, 0, stream>>>((unsigned*)zbeg, zwords);

    // 1) precision casts
    k_cast8<<<cdiv((long long)n * IN_F / 8, 256), 256, 0, stream>>>(x, xb, (long long)n * IN_F / 8);
    k_castT<<<cdiv(HID_F * IN_F, 256), 256, 0, stream>>>(W1, W1t, HID_F, IN_F);
    k_castT<<<cdiv(OUT_F * HID_F, 256), 256, 0, stream>>>(W2, W2t, OUT_F, HID_F);

    // 2) tiny precomputes
    k_prep<<<1, RDIM * HEADS, 0, stream>>>(We1, att_e1, rel_emb, rel_alpha);
    k_cnt<<<cdiv(E, 256), 256, 0, stream>>>(dstE, etype, cnt, E);

    // 3) GEMM1: h1 = x @ W1  (bf16 WMMA, async-LDS pipeline, f32 accumulate)
    {
        dim3 grid(cdiv(n, 64), HID_F / 128);
        k_gemm_bf16<<<grid, 256, 0, stream>>>(xb, W1t, h1, n, HID_F, IN_F);
    }

    // 4) layer-1 attention
    k_nodeatt1<<<cdiv((long long)n * HEADS, 256), 256, 0, stream>>>(h1, att_src1, att_dst1,
                                                                    cnt, rel_alpha, al_s1,
                                                                    al_d1, sa1, n);
    k_amax1<<<cdiv((long long)Etot * HEADS, 256), 256, 0, stream>>>(srcE, dstE, etype, al_s1,
                                                                     al_d1, rel_alpha, sa1,
                                                                     ex1, mx1, E, Etot);
    k_aexp1<<<cdiv((long long)Etot * HEADS, 256), 256, 0, stream>>>(dstE, ex1, mx1, den1, E, Etot);
    k_agg1<<<cdiv(Etot, 8), 256, 0, stream>>>(srcE, dstE, h1, ex1, den1, out1, E, Etot);

    // 5) bias + LN + ELU -> bf16 activations
    k_lnelu1<<<cdiv(n, 8), 256, 0, stream>>>(out1, b1, g1, beta1, hact, n);

    // 6) GEMM2: h2 = hact @ W2
    {
        dim3 grid(cdiv(n, 64), OUT_F / 128);
        k_gemm_bf16<<<grid, 256, 0, stream>>>(hact, W2t, h2, n, OUT_F, HID_F);
    }

    // 7) layer-2 attention (single head)
    k_nodeatt2<<<cdiv(n, 8), 256, 0, stream>>>(h2, att_src2, att_dst2, al_s2, al_d2, n);
    k_amax2<<<cdiv(Etot, 256), 256, 0, stream>>>(srcE, dstE, al_s2, al_d2, ex2, mx2, E, Etot);
    k_aexp2<<<cdiv(Etot, 256), 256, 0, stream>>>(dstE, ex2, mx2, den2, E, Etot);
    k_agg2<<<cdiv(Etot, 8), 256, 0, stream>>>(srcE, dstE, h2, ex2, den2, out2, E, Etot);

    // 8) final bias + LN -> d_out
    k_ln2<<<cdiv(n, 8), 256, 0, stream>>>(out2, b2, g2, beta2, (float*)d_out, n);
}
